// transformer_pointer_network_decoder_64690797412769
// MI455X (gfx1250) — matrix operand order repaired and checked by reading, not compiled, not run on hardware
//
#include <hip/hip_runtime.h>
#include <hip/hip_bf16.h>

typedef __attribute__((ext_vector_type(16))) _Float16 v16h;
typedef __attribute__((ext_vector_type(8)))  _Float16 v8h;
typedef __attribute__((ext_vector_type(8)))  float    v8f;
typedef __attribute__((ext_vector_type(4)))  float    v4f;

namespace {
constexpr int DIM = 512;   // model dim
constexpr int NH  = 8;     // heads
constexpr int DH  = 64;    // head dim
constexpr int FF  = 2048;  // ffn dim
constexpr int NL  = 6;     // layers
constexpr int NB  = 32;    // batch
constexpr int NT  = 20;    // steps
constexpr int NA  = 21;    // actions
constexpr int H3  = 768;   // tgt hidden
constexpr int NM  = 20;    // memory len
constexpr float LN_EPS = 1e-5f;
}

// ---------------------------------------------------------------------------
// fp32 -> f16 weight conversion (one-time per call; weights then L2-resident)
// ---------------------------------------------------------------------------
__global__ void f32_to_f16_kernel(const float* __restrict__ in,
                                  _Float16* __restrict__ out, int n) {
  for (int i = blockIdx.x * blockDim.x + threadIdx.x; i < n;
       i += gridDim.x * blockDim.x)
    out[i] = (_Float16)in[i];
}

// Assemble WMMA A fragment (16-bit A 16x32 layout) from 16 fp32 values.
__device__ __forceinline__ v16h pack_a(v4f f0, v4f f1, v4f f2, v4f f3) {
  v16h a;
#pragma unroll
  for (int i = 0; i < 4; i++) {
    a[i]      = (_Float16)f0[i];
    a[4 + i]  = (_Float16)f1[i];
    a[8 + i]  = (_Float16)f2[i];
    a[12 + i] = (_Float16)f3[i];
  }
  return a;
}

// ---------------------------------------------------------------------------
// WMMA GEMM:  C[M,N] = A[M,K] (fp32, cvt->f16 in-reg) * W[N,K]^T (f16) + bias
// One wave computes a 16x64 output strip = 4 N-tiles:
//   - A fragment loaded/converted once per k-step, reused 4x
//   - 4 independent accumulator chains
//   - software-pipelined: next-iter loads issue before current WMMAs,
//     conversion happens after them, buffers are disjoint (no WAR nops)
// M mult of 16, N mult of 64, K mult of 32 -> no guards, EXEC all ones.
// ---------------------------------------------------------------------------
__global__ __launch_bounds__(32) void gemm_wmma_kernel(
    const float* __restrict__ Am, const _Float16* __restrict__ Wm,
    const float* __restrict__ bias, float* __restrict__ Cm,
    int M, int N, int K, int relu) {
  const int lane = threadIdx.x;
  const int rr   = lane & 15;
  const int hi   = lane >> 4;
  const int tm   = blockIdx.x;
  const int tn0  = blockIdx.y * 4;
  const int m    = tm * 16 + rr;

  v8f acc0 = {0.f, 0.f, 0.f, 0.f, 0.f, 0.f, 0.f, 0.f};
  v8f acc1 = acc0, acc2 = acc0, acc3 = acc0;

  const float*    arow = Am + (size_t)m * K;
  const _Float16* br0  = Wm + (size_t)((tn0 + 0) * 16 + rr) * K;
  const _Float16* br1  = Wm + (size_t)((tn0 + 1) * 16 + rr) * K;
  const _Float16* br2  = Wm + (size_t)((tn0 + 2) * 16 + rr) * K;
  const _Float16* br3  = Wm + (size_t)((tn0 + 3) * 16 + rr) * K;

  // --- prologue: fragments for k0 = 0 ------------------------------------
  v4f f0 = ((const v4f*)(arow + hi * 8))[0];
  v4f f1 = ((const v4f*)(arow + hi * 8))[1];
  v4f f2 = ((const v4f*)(arow + 16 + hi * 8))[0];
  v4f f3 = ((const v4f*)(arow + 16 + hi * 8))[1];
  v16h b0 = __builtin_shufflevector(*(const v8h*)(br0 + hi * 8), *(const v8h*)(br0 + 16 + hi * 8), 0, 1, 2, 3, 4, 5, 6, 7, 8, 9, 10, 11, 12, 13, 14, 15);
  v16h b1 = __builtin_shufflevector(*(const v8h*)(br1 + hi * 8), *(const v8h*)(br1 + 16 + hi * 8), 0, 1, 2, 3, 4, 5, 6, 7, 8, 9, 10, 11, 12, 13, 14, 15);
  v16h b2 = __builtin_shufflevector(*(const v8h*)(br2 + hi * 8), *(const v8h*)(br2 + 16 + hi * 8), 0, 1, 2, 3, 4, 5, 6, 7, 8, 9, 10, 11, 12, 13, 14, 15);
  v16h b3 = __builtin_shufflevector(*(const v8h*)(br3 + hi * 8), *(const v8h*)(br3 + 16 + hi * 8), 0, 1, 2, 3, 4, 5, 6, 7, 8, 9, 10, 11, 12, 13, 14, 15);
  v16h a = pack_a(f0, f1, f2, f3);

  // --- steady state: prefetch k0+32, compute k0 ---------------------------
  int k0 = 0;
  for (; k0 + 32 < K; k0 += 32) {
    const int k1 = k0 + 32;
    v4f g0 = ((const v4f*)(arow + k1 + hi * 8))[0];
    v4f g1 = ((const v4f*)(arow + k1 + hi * 8))[1];
    v4f g2 = ((const v4f*)(arow + k1 + 16 + hi * 8))[0];
    v4f g3 = ((const v4f*)(arow + k1 + 16 + hi * 8))[1];
    v16h c0 = __builtin_shufflevector(*(const v8h*)(br0 + k1 + hi * 8), *(const v8h*)(br0 + k1 + 16 + hi * 8), 0, 1, 2, 3, 4, 5, 6, 7, 8, 9, 10, 11, 12, 13, 14, 15);
    v16h c1 = __builtin_shufflevector(*(const v8h*)(br1 + k1 + hi * 8), *(const v8h*)(br1 + k1 + 16 + hi * 8), 0, 1, 2, 3, 4, 5, 6, 7, 8, 9, 10, 11, 12, 13, 14, 15);
    v16h c2 = __builtin_shufflevector(*(const v8h*)(br2 + k1 + hi * 8), *(const v8h*)(br2 + k1 + 16 + hi * 8), 0, 1, 2, 3, 4, 5, 6, 7, 8, 9, 10, 11, 12, 13, 14, 15);
    v16h c3 = __builtin_shufflevector(*(const v8h*)(br3 + k1 + hi * 8), *(const v8h*)(br3 + k1 + 16 + hi * 8), 0, 1, 2, 3, 4, 5, 6, 7, 8, 9, 10, 11, 12, 13, 14, 15);
    acc0 = __builtin_amdgcn_wmma_f32_16x16x32_f16(false, a, false, b0,
                                                  (short)0, acc0, false, false);
    acc1 = __builtin_amdgcn_wmma_f32_16x16x32_f16(false, a, false, b1,
                                                  (short)0, acc1, false, false);
    acc2 = __builtin_amdgcn_wmma_f32_16x16x32_f16(false, a, false, b2,
                                                  (short)0, acc2, false, false);
    acc3 = __builtin_amdgcn_wmma_f32_16x16x32_f16(false, a, false, b3,
                                                  (short)0, acc3, false, false);
    a = pack_a(g0, g1, g2, g3);
    b0 = c0; b1 = c1; b2 = c2; b3 = c3;
  }
  // --- epilogue k-step ----------------------------------------------------
  acc0 = __builtin_amdgcn_wmma_f32_16x16x32_f16(false, a, false, b0, (short)0,
                                                acc0, false, false);
  acc1 = __builtin_amdgcn_wmma_f32_16x16x32_f16(false, a, false, b1, (short)0,
                                                acc1, false, false);
  acc2 = __builtin_amdgcn_wmma_f32_16x16x32_f16(false, a, false, b2, (short)0,
                                                acc2, false, false);
  acc3 = __builtin_amdgcn_wmma_f32_16x16x32_f16(false, a, false, b3, (short)0,
                                                acc3, false, false);

#pragma unroll
  for (int t = 0; t < 4; t++) {
    const v8f* accp = (t == 0) ? &acc0 : (t == 1) ? &acc1 : (t == 2) ? &acc2
                                                                     : &acc3;
    v8f acc = *accp;
    const int ncol = (tn0 + t) * 16 + rr;
    const float bv = bias[ncol];
#pragma unroll
    for (int v = 0; v < 8; v++) {
      int row = tm * 16 + hi * 8 + v;     // D layout: VGPR v -> M = half*8+v
      float val = acc[v] + bv;
      if (relu) val = fmaxf(val, 0.f);
      Cm[(size_t)row * N + ncol] = val;
    }
  }
}

// ---------------------------------------------------------------------------
// Self-attention (tiny: S<=20, dh=64) — fp32 in LDS, one block per (b,h)
// qkv rows: [q(512) | k(512) | v(512)]
// ---------------------------------------------------------------------------
__global__ __launch_bounds__(256) void self_attn_kernel(
    const float* __restrict__ qkv, float* __restrict__ out, int S) {
  int b = blockIdx.x / NH;
  int h = blockIdx.x % NH;
  __shared__ float qs[NT][DH], ks[NT][DH], vs[NT][DH], sc[NT][NT];
  for (int idx = threadIdx.x; idx < S * DH; idx += 256) {
    int s = idx / DH, d = idx % DH;
    const float* row = qkv + (size_t)(b * S + s) * (3 * DIM);
    qs[s][d] = row[h * DH + d];
    ks[s][d] = row[DIM + h * DH + d];
    vs[s][d] = row[2 * DIM + h * DH + d];
  }
  __syncthreads();
  for (int idx = threadIdx.x; idx < S * S; idx += 256) {
    int s = idx / S, j = idx % S;
    float acc = 0.f;
#pragma unroll
    for (int d = 0; d < DH; d++) acc += qs[s][d] * ks[j][d];
    sc[s][j] = acc * 0.125f;  // 1/sqrt(64)
  }
  __syncthreads();
  if (threadIdx.x < (unsigned)S) {
    int s = threadIdx.x;
    float mx = -1e30f;
    for (int j = 0; j < S; j++) mx = fmaxf(mx, sc[s][j]);
    float sum = 0.f;
    for (int j = 0; j < S; j++) { float e = expf(sc[s][j] - mx); sc[s][j] = e; sum += e; }
    float inv = 1.f / sum;
    for (int j = 0; j < S; j++) sc[s][j] *= inv;
  }
  __syncthreads();
  for (int idx = threadIdx.x; idx < S * DH; idx += 256) {
    int s = idx / DH, d = idx % DH;
    float acc = 0.f;
    for (int j = 0; j < S; j++) acc += sc[s][j] * vs[j][d];
    out[(size_t)(b * S + s) * DIM + h * DH + d] = acc;
  }
}

// ---------------------------------------------------------------------------
// Cross-attention: q from qc (B*S x 512), K/V precomputed per layer (B*NM x 512)
// ---------------------------------------------------------------------------
__global__ __launch_bounds__(256) void cross_attn_kernel(
    const float* __restrict__ qc, const float* __restrict__ kmem,
    const float* __restrict__ vmem, float* __restrict__ out, int S) {
  int b = blockIdx.x / NH;
  int h = blockIdx.x % NH;
  __shared__ float qs[NT][DH], ks[NM][DH], vs[NM][DH], sc[NT][NM];
  for (int idx = threadIdx.x; idx < S * DH; idx += 256) {
    int s = idx / DH, d = idx % DH;
    qs[s][d] = qc[(size_t)(b * S + s) * DIM + h * DH + d];
  }
  for (int idx = threadIdx.x; idx < NM * DH; idx += 256) {
    int j = idx / DH, d = idx % DH;
    ks[j][d] = kmem[(size_t)(b * NM + j) * DIM + h * DH + d];
    vs[j][d] = vmem[(size_t)(b * NM + j) * DIM + h * DH + d];
  }
  __syncthreads();
  for (int idx = threadIdx.x; idx < S * NM; idx += 256) {
    int s = idx / NM, j = idx % NM;
    float acc = 0.f;
#pragma unroll
    for (int d = 0; d < DH; d++) acc += qs[s][d] * ks[j][d];
    sc[s][j] = acc * 0.125f;
  }
  __syncthreads();
  if (threadIdx.x < (unsigned)S) {
    int s = threadIdx.x;
    float mx = -1e30f;
    for (int j = 0; j < NM; j++) mx = fmaxf(mx, sc[s][j]);
    float sum = 0.f;
    for (int j = 0; j < NM; j++) { float e = expf(sc[s][j] - mx); sc[s][j] = e; sum += e; }
    float inv = 1.f / sum;
    for (int j = 0; j < NM; j++) sc[s][j] *= inv;
  }
  __syncthreads();
  for (int idx = threadIdx.x; idx < S * DH; idx += 256) {
    int s = idx / DH, d = idx % DH;
    float acc = 0.f;
    for (int j = 0; j < NM; j++) acc += sc[s][j] * vs[j][d];
    out[(size_t)(b * S + s) * DIM + h * DH + d] = acc;
  }
}

// ---------------------------------------------------------------------------
// x = LayerNorm(x + proj) * g + b, in place. One block per row, 256 thr.
// ---------------------------------------------------------------------------
__global__ __launch_bounds__(256) void add_ln_kernel(
    float* __restrict__ x, const float* __restrict__ proj,
    const float* __restrict__ g, const float* __restrict__ bb) {
  int row = blockIdx.x, t = threadIdx.x;
  __shared__ float s1[256], s2[256];
  __shared__ float mean, inv;
  float* xr = x + (size_t)row * DIM;
  const float* pr = proj + (size_t)row * DIM;
  float a0 = xr[t] + pr[t];
  float a1 = xr[t + 256] + pr[t + 256];
  s1[t] = a0 + a1;
  s2[t] = a0 * a0 + a1 * a1;
  __syncthreads();
  for (int off = 128; off > 0; off >>= 1) {
    if (t < off) { s1[t] += s1[t + off]; s2[t] += s2[t + off]; }
    __syncthreads();
  }
  if (t == 0) {
    float m = s1[0] * (1.f / DIM);
    float var = s2[0] * (1.f / DIM) - m * m;
    mean = m;
    inv = rsqrtf(var + LN_EPS);
  }
  __syncthreads();
  xr[t]       = (a0 - mean) * inv * g[t]       + bb[t];
  xr[t + 256] = (a1 - mean) * inv * g[t + 256] + bb[t + 256];
}

// Final LN on last position only -> hlast[b, 512]
__global__ __launch_bounds__(256) void lnf_last_kernel(
    const float* __restrict__ x, const float* __restrict__ g,
    const float* __restrict__ bb, float* __restrict__ out, int S) {
  int b = blockIdx.x, t = threadIdx.x;
  __shared__ float s1[256], s2[256];
  __shared__ float mean, inv;
  const float* xr = x + (size_t)(b * S + S - 1) * DIM;
  float a0 = xr[t], a1 = xr[t + 256];
  s1[t] = a0 + a1;
  s2[t] = a0 * a0 + a1 * a1;
  __syncthreads();
  for (int off = 128; off > 0; off >>= 1) {
    if (t < off) { s1[t] += s1[t + off]; s2[t] += s2[t + off]; }
    __syncthreads();
  }
  if (t == 0) {
    float m = s1[0] * (1.f / DIM);
    float var = s2[0] * (1.f / DIM) - m * m;
    mean = m;
    inv = rsqrtf(var + LN_EPS);
  }
  __syncthreads();
  out[(size_t)b * DIM + t]       = (a0 - mean) * inv * g[t]       + bb[t];
  out[(size_t)b * DIM + t + 256] = (a1 - mean) * inv * g[t + 256] + bb[t + 256];
}

// ---------------------------------------------------------------------------
// Sequence construction (fully static given action_list)
// ---------------------------------------------------------------------------
__global__ void build_concat_kernel(const float* __restrict__ emb,
                                    const float* __restrict__ initd,
                                    float* __restrict__ concat) {
  int n = NB * (NT + 1) * DIM;
  for (int i = blockIdx.x * blockDim.x + threadIdx.x; i < n;
       i += gridDim.x * blockDim.x) {
    int d = i % DIM, rest = i / DIM;
    int pos = rest % (NT + 1), b = rest / (NT + 1);
    concat[i] = (pos == 0) ? initd[d]
                           : emb[((size_t)b * NT + (pos - 1)) * DIM + d];
  }
}

__global__ void build_seq_kernel(const float* __restrict__ concat,
                                 const float* __restrict__ initd,
                                 const int* __restrict__ act,
                                 float* __restrict__ seq) {
  int n = NB * NT * DIM;
  for (int i = blockIdx.x * blockDim.x + threadIdx.x; i < n;
       i += gridDim.x * blockDim.x) {
    int d = i % DIM, rest = i / DIM;
    int pos = rest % NT, b = rest / NT;
    if (pos == 0) {
      seq[i] = initd[d];
    } else {
      int a = act[b * NT + pos - 1];
      seq[i] = concat[((size_t)b * (NT + 1) + a) * DIM + d];
    }
  }
}

__global__ void copy_x_kernel(float* __restrict__ x,
                              const float* __restrict__ seq, int S) {
  int n = NB * S * DIM;
  for (int i = blockIdx.x * blockDim.x + threadIdx.x; i < n;
       i += gridDim.x * blockDim.x) {
    int d = i % DIM, rest = i / DIM;
    int s = rest % S, b = rest / S;
    x[i] = seq[((size_t)b * NT + s) * DIM + d];
  }
}

// ---------------------------------------------------------------------------
// av[b, 21] = hlast[b] @ aff_W^T + aff_b  (tiny; one wave per batch row)
// ---------------------------------------------------------------------------
__global__ __launch_bounds__(32) void aff_kernel(
    const float* __restrict__ h, const float* __restrict__ W,
    const float* __restrict__ bias, float* __restrict__ av) {
  int b = blockIdx.x, t = threadIdx.x;
  if (t < NA) {
    float acc = bias[t];
    const float* hr = h + (size_t)b * DIM;
    const float* wr = W + (size_t)t * DIM;
    for (int d = 0; d < DIM; d++) acc += hr[d] * wr[d];
    av[b * NA + t] = acc;
  }
}

// ---------------------------------------------------------------------------
// Per-step masked log-softmax / entropy / pointer bookkeeping (B=32 lanes)
// ---------------------------------------------------------------------------
__global__ __launch_bounds__(32) void step_kernel(
    const float* __restrict__ av, const int* __restrict__ act, int i,
    float* mask, float* term, float* ljp, float* ces, float* lastp,
    float* prob, float* __restrict__ out) {
  int b = threadIdx.x;
  float mx = -1e30f;
  for (int a = 0; a < NA; a++) {
    bool neg = (mask[b * NA + a] != 0.f) || (i == 0 && a == 0);
    if (!neg) mx = fmaxf(mx, av[b * NA + a]);
  }
  float sum = 0.f;
  for (int a = 0; a < NA; a++) {
    bool neg = (mask[b * NA + a] != 0.f) || (i == 0 && a == 0);
    if (!neg) sum += expf(av[b * NA + a] - mx);
  }
  float lse = mx + logf(sum);
  float ent = 0.f;
  for (int a = 0; a < NA; a++) {
    bool neg = (mask[b * NA + a] != 0.f) || (i == 0 && a == 0);
    if (!neg) {
      float lp = av[b * NA + a] - lse;
      ent -= expf(lp) * lp;
    }
  }
  if (i > 0) lastp[b] *= prob[b];
  bool tm = term[b] != 0.f;
  ces[b] += (tm ? 0.f : ent) * lastp[b];
  int idx = act[b * NT + i];
  bool negIdx = (mask[b * NA + idx] != 0.f) || (i == 0 && idx == 0);
  float pi = negIdx ? 0.f : expf(av[b * NA + idx] - lse);
  ljp[b] += tm ? 0.f : logf(pi + 1e-7f);
  mask[b * NA + idx] = 1.f;
  term[b] = (idx == 0) ? 1.f : 0.f;
  prob[b] = pi;
  out[b] = ljp[b];
  out[NB + b] = ces[b];
}

__global__ void init_state_kernel(float* mask, float* term, float* ljp,
                                  float* ces, float* lastp, float* prob) {
  int t = blockIdx.x * blockDim.x + threadIdx.x;
  if (t < NB * NA) mask[t] = 0.f;
  if (t < NB) {
    term[t] = 0.f; ljp[t] = 0.f; ces[t] = 0.f;
    lastp[t] = 1.f; prob[t] = 1.f;
  }
}

// ---------------------------------------------------------------------------
extern "C" void kernel_launch(void* const* d_in, const int* in_sizes, int n_in,
                              void* d_out, int out_size, void* d_ws,
                              size_t ws_size, hipStream_t stream) {
  (void)in_sizes; (void)n_in; (void)out_size; (void)ws_size;
  const float* tgt      = (const float*)d_in[0];
  const float* memory   = (const float*)d_in[1];
  const float* Wqkv_s   = (const float*)d_in[2];
  const float* bqkv_s   = (const float*)d_in[3];
  const float* Wo_s     = (const float*)d_in[4];
  const float* bo_s     = (const float*)d_in[5];
  const float* Wqkv_c   = (const float*)d_in[6];
  const float* bqkv_c   = (const float*)d_in[7];
  const float* Wo_c     = (const float*)d_in[8];
  const float* bo_c     = (const float*)d_in[9];
  const float* W1       = (const float*)d_in[10];
  const float* b1       = (const float*)d_in[11];
  const float* W2       = (const float*)d_in[12];
  const float* b2       = (const float*)d_in[13];
  const float* ln1_g    = (const float*)d_in[14];
  const float* ln1_b    = (const float*)d_in[15];
  const float* ln2_g    = (const float*)d_in[16];
  const float* ln2_b    = (const float*)d_in[17];
  const float* ln3_g    = (const float*)d_in[18];
  const float* ln3_b    = (const float*)d_in[19];
  const float* lnf_g    = (const float*)d_in[20];
  const float* lnf_b    = (const float*)d_in[21];
  const float* emb_W    = (const float*)d_in[22];
  const float* emb_b    = (const float*)d_in[23];
  const float* aff_W    = (const float*)d_in[24];
  const float* aff_b    = (const float*)d_in[25];
  const float* init_dec = (const float*)d_in[26];
  const int*   act      = (const int*)d_in[27];

  // --- bump allocator over d_ws ------------------------------------------
  char* p = (char*)d_ws;
  auto alloc = [&](size_t bytes) -> void* {
    void* r = (void*)p;
    p += (bytes + 255) & ~(size_t)255;
    return r;
  };

  _Float16* wqkv_s16 = (_Float16*)alloc((size_t)NL * 3 * DIM * DIM * 2);
  _Float16* wo_s16   = (_Float16*)alloc((size_t)NL * DIM * DIM * 2);
  _Float16* wqkv_c16 = (_Float16*)alloc((size_t)NL * 3 * DIM * DIM * 2);
  _Float16* wo_c16   = (_Float16*)alloc((size_t)NL * DIM * DIM * 2);
  _Float16* w1_16    = (_Float16*)alloc((size_t)NL * FF * DIM * 2);
  _Float16* w2_16    = (_Float16*)alloc((size_t)NL * DIM * FF * 2);
  _Float16* embW16   = (_Float16*)alloc((size_t)DIM * H3 * 2);

  float* memK   = (float*)alloc((size_t)NL * NB * NM * DIM * 4);
  float* memV   = (float*)alloc((size_t)NL * NB * NM * DIM * 4);
  float* emb    = (float*)alloc((size_t)NB * NT * DIM * 4);
  float* concat = (float*)alloc((size_t)NB * (NT + 1) * DIM * 4);
  float* seq    = (float*)alloc((size_t)NB * NT * DIM * 4);
  float* x      = (float*)alloc((size_t)NB * NT * DIM * 4);
  float* qkv    = (float*)alloc((size_t)NB * NT * 3 * DIM * 4);
  float* sattn  = (float*)alloc((size_t)NB * NT * DIM * 4);
  float* proj   = (float*)alloc((size_t)NB * NT * DIM * 4);
  float* qc     = (float*)alloc((size_t)NB * NT * DIM * 4);
  float* cattn  = (float*)alloc((size_t)NB * NT * DIM * 4);
  float* h1     = (float*)alloc((size_t)NB * NT * FF * 4);
  float* hlast  = (float*)alloc((size_t)NB * DIM * 4);
  float* av     = (float*)alloc((size_t)NB * NA * 4);
  float* maskS  = (float*)alloc((size_t)NB * NA * 4);
  float* termS  = (float*)alloc((size_t)NB * 4);
  float* ljpS   = (float*)alloc((size_t)NB * 4);
  float* cesS   = (float*)alloc((size_t)NB * 4);
  float* lastpS = (float*)alloc((size_t)NB * 4);
  float* probS  = (float*)alloc((size_t)NB * 4);

  auto cvt = [&](const float* src, _Float16* dst, size_t n) {
    f32_to_f16_kernel<<<dim3(1024), dim3(256), 0, stream>>>(src, dst, (int)n);
  };
  auto gemm = [&](const float* Am, const _Float16* Wm, const float* bias,
                  float* Cm, int M, int N, int K, int relu) {
    gemm_wmma_kernel<<<dim3(M / 16, N / 64), dim3(32), 0, stream>>>(
        Am, Wm, bias, Cm, M, N, K, relu);
  };

  // --- one-time setup -----------------------------------------------------
  cvt(Wqkv_s, wqkv_s16, (size_t)NL * 3 * DIM * DIM);
  cvt(Wo_s,   wo_s16,   (size_t)NL * DIM * DIM);
  cvt(Wqkv_c, wqkv_c16, (size_t)NL * 3 * DIM * DIM);
  cvt(Wo_c,   wo_c16,   (size_t)NL * DIM * DIM);
  cvt(W1,     w1_16,    (size_t)NL * FF * DIM);
  cvt(W2,     w2_16,    (size_t)NL * DIM * FF);
  cvt(emb_W,  embW16,   (size_t)DIM * H3);

  // emb = relu(tgt @ emb_W.T + emb_b) : (B*T x 768) x (512 x 768)^T
  gemm(tgt, embW16, emb_b, emb, NB * NT, DIM, H3, 1);
  build_concat_kernel<<<dim3(128), dim3(256), 0, stream>>>(emb, init_dec, concat);
  build_seq_kernel<<<dim3(128), dim3(256), 0, stream>>>(concat, init_dec, act, seq);

  // Cross-attn K/V for memory, per layer (memory is constant across steps)
  for (int l = 0; l < NL; l++) {
    gemm(memory, wqkv_c16 + (size_t)l * 3 * DIM * DIM + (size_t)DIM * DIM,
         bqkv_c + l * 3 * DIM + DIM, memK + (size_t)l * NB * NM * DIM,
         NB * NM, DIM, DIM, 0);
    gemm(memory, wqkv_c16 + (size_t)l * 3 * DIM * DIM + (size_t)2 * DIM * DIM,
         bqkv_c + l * 3 * DIM + 2 * DIM, memV + (size_t)l * NB * NM * DIM,
         NB * NM, DIM, DIM, 0);
  }
  init_state_kernel<<<dim3(3), dim3(256), 0, stream>>>(maskS, termS, ljpS,
                                                       cesS, lastpS, probS);

  // --- sequential decode steps -------------------------------------------
  for (int i = 0; i < NT; i++) {
    int S = i + 1;
    int rows = NB * S;
    copy_x_kernel<<<dim3(256), dim3(256), 0, stream>>>(x, seq, S);
    for (int l = 0; l < NL; l++) {
      // self-attention block
      gemm(x, wqkv_s16 + (size_t)l * 3 * DIM * DIM, bqkv_s + l * 3 * DIM,
           qkv, rows, 3 * DIM, DIM, 0);
      self_attn_kernel<<<dim3(NB * NH), dim3(256), 0, stream>>>(qkv, sattn, S);
      gemm(sattn, wo_s16 + (size_t)l * DIM * DIM, bo_s + l * DIM, proj,
           rows, DIM, DIM, 0);
      add_ln_kernel<<<dim3(rows), dim3(256), 0, stream>>>(
          x, proj, ln1_g + l * DIM, ln1_b + l * DIM);
      // cross-attention block
      gemm(x, wqkv_c16 + (size_t)l * 3 * DIM * DIM, bqkv_c + l * 3 * DIM,
           qc, rows, DIM, DIM, 0);
      cross_attn_kernel<<<dim3(NB * NH), dim3(256), 0, stream>>>(
          qc, memK + (size_t)l * NB * NM * DIM,
          memV + (size_t)l * NB * NM * DIM, cattn, S);
      gemm(cattn, wo_c16 + (size_t)l * DIM * DIM, bo_c + l * DIM, proj,
           rows, DIM, DIM, 0);
      add_ln_kernel<<<dim3(rows), dim3(256), 0, stream>>>(
          x, proj, ln2_g + l * DIM, ln2_b + l * DIM);
      // FFN block
      gemm(x, w1_16 + (size_t)l * FF * DIM, b1 + l * FF, h1, rows, FF, DIM, 1);
      gemm(h1, w2_16 + (size_t)l * DIM * FF, b2 + l * DIM, proj, rows, DIM,
           FF, 0);
      add_ln_kernel<<<dim3(rows), dim3(256), 0, stream>>>(
          x, proj, ln3_g + l * DIM, ln3_b + l * DIM);
    }
    lnf_last_kernel<<<dim3(NB), dim3(256), 0, stream>>>(x, lnf_g, lnf_b,
                                                        hlast, S);
    aff_kernel<<<dim3(NB), dim3(32), 0, stream>>>(hlast, aff_W, aff_b, av);
    step_kernel<<<dim3(1), dim3(32), 0, stream>>>(
        av, act, i, maskS, termS, ljpS, cesS, lastpS, probS, (float*)d_out);
  }
}
